// EchoStateNetwork_85512798863668
// MI455X (gfx1250) — compile-verified, matrix operand order checked by reading, not run on hardware
//
#include <hip/hip_runtime.h>
#include <hip/hip_bf16.h>

// Echo State Network forward for MI455X (gfx1250, wave32, WMMA bf16->f32).
//
// inputs: x[32,1024,128] f32, W_res[1024,1024] f32, W_in[1024,128] f32,
//         W_out[128,1024] f32, b_out[128] f32
// output: [32,1024,128] f32
//
// Recurrence: 32 persistent blocks x 256 threads (8 waves = 2 N-tiles x 4
// K-quarters). Block's 32x1024 W_res slice is LDS-resident (padded rows,
// conflict-free ds_load_b128). Input drive u_{t+1} is computed during step t
// with a clamped index (branch-free); leaky f32 state h is register-resident
// in K-quarter-0 waves. Per-step cross-wave K reduction goes through 12 KB of
// LDS; steps are separated by an agent-scope spin barrier. A-fragment loads
// are software-pipelined. __launch_bounds__(256,1) gives the allocator a real
// register budget (no scratch spills).

#define B_   32
#define T_   1024
#define D_   128
#define R_   1024
#define O_   128
#define LEAK 0.3f

#define NBLK 32      // persistent blocks; each owns 32 reservoir columns
#define ROWP 1032    // padded LDS row stride (bf16): 2064B = 516 dwords -> stride-4 banks

typedef __attribute__((ext_vector_type(16))) __bf16 v16bf;
typedef __attribute__((ext_vector_type(8)))  __bf16 v8bf;
typedef __attribute__((ext_vector_type(8)))  float  v8f;

// Fragment layout per CDNA5 ISA 7.12.2 (16-bit A 16x32 / B 32x16):
// lane row = l15, elems 0..7 = K[kb..kb+7], elems 8..15 = K[kb+16..kb+23],
// kb = 8*(lane>>4) + 32*kstep.  Two contiguous 16B loads per fragment.
__device__ __forceinline__ v16bf frag_ld(const __hip_bfloat16* __restrict__ p) {
  v8bf lo = *(const v8bf*)(p);
  v8bf hi = *(const v8bf*)(p + 16);
  return __builtin_shufflevector(lo, hi, 0, 1, 2, 3, 4, 5, 6, 7,
                                         8, 9, 10, 11, 12, 13, 14, 15);
}

__device__ __forceinline__ v16bf frag_ld_lds(const __bf16* smem, int row, int kb) {
  v8bf lo = *(const v8bf*)&smem[row * ROWP + kb];
  v8bf hi = *(const v8bf*)&smem[row * ROWP + kb + 16];
  return __builtin_shufflevector(lo, hi, 0, 1, 2, 3, 4, 5, 6, 7,
                                         8, 9, 10, 11, 12, 13, 14, 15);
}

__device__ __forceinline__ v8f wmma_bf16(v16bf a, v16bf b, v8f c) {
  return __builtin_amdgcn_wmma_f32_16x16x32_bf16(
      /*neg_a=*/false, a, /*neg_b=*/false, b,
      /*c_mod=*/(short)0, c, /*reuse_a=*/false, /*reuse_b=*/false);
}

// ---------------------------------------------------------------------------
__global__ void __launch_bounds__(256) cvt_f32_bf16(const float* __restrict__ src,
                                                    __hip_bfloat16* __restrict__ dst,
                                                    int n) {
  for (int i = blockIdx.x * blockDim.x + threadIdx.x; i < n;
       i += gridDim.x * blockDim.x)
    dst[i] = __float2bfloat16(src[i]);
}

// ---------------------------------------------------------------------------
__global__ void __launch_bounds__(256, 1)
esn_recurrence(const __hip_bfloat16* __restrict__ xb,    // [32,1024,128]
               const __hip_bfloat16* __restrict__ winb,  // [1024,128]
               const __hip_bfloat16* __restrict__ wresb, // [1024,1024]
               __hip_bfloat16* __restrict__ st,          // [32,1024,1024]
               unsigned* __restrict__ bar) {
  extern __shared__ __bf16 smem[];                 // [32][ROWP] W_res slice
  float* red = (float*)(smem + 32 * ROWP);         // 12KB partial-sum scratch

  const int nb0 = blockIdx.x * 32;                 // first reservoir column owned

  // stage W_res slice (rows nb0..nb0+31, all K) into LDS once
  for (int i = threadIdx.x; i < 32 * 128; i += blockDim.x) {
    int r = i >> 7;
    int c = (i & 127) * 8;
    *(uint4*)&smem[r * ROWP + c] =
        *(const uint4*)&wresb[(size_t)(nb0 + r) * R_ + c];
  }
  __syncthreads();

  const int lane  = threadIdx.x & 31;
  const int wave  = threadIdx.x >> 5;
  const int ntile = wave & 1;               // 0..1
  const int kq    = wave >> 1;              // 0..3 K-quarter
  const int l15   = lane & 15;
  const int klane = (lane >> 4) << 3;       // 0 or 8
  const int half8 = klane;                  // C-row offset for this lane half
  const int nlane = nb0 + ntile * 16 + l15; // global reservoir column
  const int nloc  = ntile * 16 + l15;       // LDS row

  // A-fragment row pointers: m-tile 0 = batch rows 0..15, m-tile 1 = 16..31
  const __hip_bfloat16* xrow0 = xb + (size_t)l15 * T_ * D_;
  const __hip_bfloat16* xrow1 = xb + (size_t)(16 + l15) * T_ * D_;
  const __hip_bfloat16* srow0 = st + (size_t)l15 * T_ * R_;
  const __hip_bfloat16* srow1 = st + (size_t)(16 + l15) * T_ * R_;

  // W_in fragment (loop-invariant; this wave covers input dims [kq*32,kq*32+32))
  const int   kbin = (kq << 5) + klane;
  const v16bf bw   = frag_ld(winb + (size_t)nlane * D_ + kbin);
  const int   kb0  = (kq << 8) + klane;     // recurrence K base for this wave

  // persistent leaky f32 state (valid in kq==0 waves)
  v8f h0 = {}, h1 = {};

  // input drive for t = 0
  v8f u0 = {}, u1 = {};
  {
    v16bf a0 = frag_ld(xrow0 + kbin);
    v16bf a1 = frag_ld(xrow1 + kbin);
    u0 = wmma_bf16(a0, bw, u0);
    u1 = wmma_bf16(a1, bw, u1);
  }

  for (int t = 0; t < T_; ++t) {
    v8f c0 = u0, c1 = u1;

    // branch-free prefetch of x for step t+1 (index clamped; last step's
    // extra drive is computed but unused, keeping the loop body copy-free)
    const int tn = (t + 1 < T_) ? (t + 1) : t;
    v16bf xa0 = frag_ld(xrow0 + (size_t)tn * D_ + kbin);
    v16bf xa1 = frag_ld(xrow1 + (size_t)tn * D_ + kbin);

    // recurrence: K in [kq*256, +256) -> 8 k-steps, loads software-pipelined
    if (t > 0) {
      const __hip_bfloat16* ap0 = srow0 + (size_t)(t - 1) * R_;
      const __hip_bfloat16* ap1 = srow1 + (size_t)(t - 1) * R_;
      v16bf a0 = frag_ld(ap0 + kb0);
      v16bf a1 = frag_ld(ap1 + kb0);
      v16bf b  = frag_ld_lds(smem, nloc, kb0);
#pragma unroll
      for (int ks = 0; ks < 8; ++ks) {
        v16bf a0n = a0, a1n = a1, bn = b;
        if (ks < 7) {
          int kb = kb0 + (ks + 1) * 32;
          a0n = frag_ld(ap0 + kb);
          a1n = frag_ld(ap1 + kb);
          bn  = frag_ld_lds(smem, nloc, kb);
        }
        c0 = wmma_bf16(a0, b, c0);
        c1 = wmma_bf16(a1, b, c1);
        a0 = a0n; a1 = a1n; b = bn;
      }
    }

    // input drive for t+1 (off the critical path; overlaps the reduction)
    u0 = wmma_bf16(xa0, bw, (v8f){});
    u1 = wmma_bf16(xa1, bw, (v8f){});

    // cross-wave K reduction through LDS: kq 1..3 publish, kq 0 accumulates
    if (kq != 0) {
      *(v8f*)&red[(((ntile * 2 + 0) * 3 + (kq - 1)) << 8) + lane * 8] = c0;
      *(v8f*)&red[(((ntile * 2 + 1) * 3 + (kq - 1)) << 8) + lane * 8] = c1;
    }
    __syncthreads();
    if (kq == 0) {
#pragma unroll
      for (int p = 0; p < 3; ++p) {
        c0 += *(const v8f*)&red[(((ntile * 2 + 0) * 3 + p) << 8) + lane * 8];
        c1 += *(const v8f*)&red[(((ntile * 2 + 1) * 3 + p) << 8) + lane * 8];
      }
      // leaky tanh update in registers; publish bf16 state for the GEMMs
#pragma unroll
      for (int g = 0; g < 8; ++g) {
        h0[g] = (1.0f - LEAK) * h0[g] + LEAK * tanhf(c0[g]);
        h1[g] = (1.0f - LEAK) * h1[g] + LEAK * tanhf(c1[g]);
      }
#pragma unroll
      for (int g = 0; g < 8; ++g) {
        int m0 = half8 + g;
        int m1 = 16 + half8 + g;
        st[((size_t)m0 * T_ + t) * R_ + nlane] = __float2bfloat16(h0[g]);
        st[((size_t)m1 * T_ + t) * R_ + nlane] = __float2bfloat16(h1[g]);
      }
    }

    // grid-wide step barrier (monotonic counter, agent-scope fences)
    __builtin_amdgcn_fence(__ATOMIC_RELEASE, "agent");
    __syncthreads();
    if (threadIdx.x == 0) {
      __hip_atomic_fetch_add(bar, 1u, __ATOMIC_RELEASE, __HIP_MEMORY_SCOPE_AGENT);
      unsigned tgt = (unsigned)(t + 1) * NBLK;
      while (__hip_atomic_load(bar, __ATOMIC_ACQUIRE, __HIP_MEMORY_SCOPE_AGENT) < tgt)
        __builtin_amdgcn_s_sleep(1);
    }
    __syncthreads();
    __builtin_amdgcn_fence(__ATOMIC_ACQUIRE, "agent");
  }
}

// ---------------------------------------------------------------------------
// Output projection: out[m,o] = states[m,:] @ W_out[o,:] + b_out[o]
// M=32768, N=128, K=1024. Each wave: 1 M-tile x 4 N-tiles (A fragment reused
// for 4 WMMAs). 4096 waves -> 512 blocks x 8 waves.
// ---------------------------------------------------------------------------
__global__ void __launch_bounds__(256)
esn_outproj(const __hip_bfloat16* __restrict__ st,
            const __hip_bfloat16* __restrict__ woutb,
            const float* __restrict__ bout,
            float* __restrict__ out) {
  const int lane  = threadIdx.x & 31;
  const int wave  = threadIdx.x >> 5;
  const int gw    = blockIdx.x * 8 + wave;
  const int mtile = gw >> 1;                // 0..2047
  const int nq    = gw & 1;                 // N-tiles nq*4 .. nq*4+3
  const int l15   = lane & 15;
  const int klane = (lane >> 4) << 3;

  const __hip_bfloat16* arow = st + (size_t)(mtile * 16 + l15) * R_;
  const __hip_bfloat16* b0 = woutb + (size_t)((nq * 4 + 0) * 16 + l15) * R_;
  const __hip_bfloat16* b1 = woutb + (size_t)((nq * 4 + 1) * 16 + l15) * R_;
  const __hip_bfloat16* b2 = woutb + (size_t)((nq * 4 + 2) * 16 + l15) * R_;
  const __hip_bfloat16* b3 = woutb + (size_t)((nq * 4 + 3) * 16 + l15) * R_;

  v8f c0 = {}, c1 = {}, c2 = {}, c3 = {};
#pragma unroll 4
  for (int ks = 0; ks < 32; ++ks) {
    int   kb = ks * 32 + klane;
    v16bf a  = frag_ld(arow + kb);
    c0 = wmma_bf16(a, frag_ld(b0 + kb), c0);
    c1 = wmma_bf16(a, frag_ld(b1 + kb), c1);
    c2 = wmma_bf16(a, frag_ld(b2 + kb), c2);
    c3 = wmma_bf16(a, frag_ld(b3 + kb), c3);
  }

  const int m0 = mtile * 16 + ((lane >> 4) << 3);
#pragma unroll
  for (int j = 0; j < 4; ++j) {
    v8f   c   = (j == 0) ? c0 : (j == 1) ? c1 : (j == 2) ? c2 : c3;
    int   n   = (nq * 4 + j) * 16 + l15;
    float bia = bout[n];
#pragma unroll
    for (int g = 0; g < 8; ++g)
      out[(size_t)(m0 + g) * O_ + n] = c[g] + bia;
  }
}

// ---------------------------------------------------------------------------
extern "C" void kernel_launch(void* const* d_in, const int* in_sizes, int n_in,
                              void* d_out, int out_size, void* d_ws, size_t ws_size,
                              hipStream_t stream) {
  const float* x    = (const float*)d_in[0];
  const float* wres = (const float*)d_in[1];
  const float* win  = (const float*)d_in[2];
  const float* wout = (const float*)d_in[3];
  const float* bout = (const float*)d_in[4];
  float* out = (float*)d_out;

  char* ws = (char*)d_ws;
  size_t off = 0;
  unsigned*       bar   = (unsigned*)(ws + off);       off += 256;
  __hip_bfloat16* wresb = (__hip_bfloat16*)(ws + off); off += (size_t)R_ * R_ * 2;
  __hip_bfloat16* winb  = (__hip_bfloat16*)(ws + off); off += (size_t)R_ * D_ * 2;
  __hip_bfloat16* woutb = (__hip_bfloat16*)(ws + off); off += (size_t)O_ * R_ * 2;
  __hip_bfloat16* xb    = (__hip_bfloat16*)(ws + off); off += (size_t)B_ * T_ * D_ * 2;
  __hip_bfloat16* st    = (__hip_bfloat16*)(ws + off); off += (size_t)B_ * T_ * R_ * 2;
  (void)ws_size; (void)in_sizes; (void)n_in; (void)out_size;

  hipMemsetAsync(bar, 0, sizeof(unsigned), stream);

  cvt_f32_bf16<<<(R_ * R_ + 255) / 256, 256, 0, stream>>>(wres, wresb, R_ * R_);
  cvt_f32_bf16<<<(R_ * D_ + 255) / 256, 256, 0, stream>>>(win, winb, R_ * D_);
  cvt_f32_bf16<<<(O_ * R_ + 255) / 256, 256, 0, stream>>>(wout, woutb, O_ * R_);
  cvt_f32_bf16<<<(B_ * T_ * D_ + 255) / 256, 256, 0, stream>>>(x, xb, B_ * T_ * D_);

  // 32 rows x ROWP bf16 (66,048 B) + 12 KB reduction scratch = 78,336 B LDS
  const size_t ldsBytes = (size_t)32 * ROWP * sizeof(__bf16) + 12 * 1024;
  esn_recurrence<<<NBLK, 256, ldsBytes, stream>>>(xb, winb, wresb, st, bar);

  esn_outproj<<<512, 256, 0, stream>>>(st, woutb, bout, out);
}
